// PortfolioGRU_36833639530615
// MI455X (gfx1250) — matrix operand order repaired and checked by reading, not compiled, not run on hardware
//
#include <hip/hip_runtime.h>
#include <hip/hip_bf16.h>
#include <math.h>

// ---------------------------------------------------------------------------
// PortfolioGRU on MI455X (gfx1250, wave32, WMMA).
//
// Heavy math (input projections + recurrent matmuls) on
// v_wmma_f32_16x16x32_f16 (f16 in, f32 accumulate). W_hh^T resident in LDS
// for the whole 250-step scan. x-proj intermediate kept in f16 (98MB).
// f16 A-tiles staged via global_load_async_to_lds_b128 (ASYNCcnt overlap);
// f32 tiles staged with b128 loads + v_cvt_pk_f16_f32 + ds_store_b128.
// ---------------------------------------------------------------------------

typedef __attribute__((ext_vector_type(16))) _Float16 v16h;
typedef __attribute__((ext_vector_type(8)))  _Float16 v8h;
typedef __attribute__((ext_vector_type(8)))  float    v8f;
typedef __attribute__((ext_vector_type(4)))  float    f32x4;

#define BATCH       512
#define SEQ         250
#define N_STOCKS    500
#define HIDDEN      128
#define G3          384                 // 3*HIDDEN
#define M_ROWS      (BATCH * SEQ)       // 128000
#define NSEL        50
#define LBc         0.0f
#define UBc         0.1f
#define LN_EPS_c    1e-5f
#define REBAL_IT    32

// ---- WMMA fragment loaders (ISA 7.12.2 layouts, wave32) --------------------
__device__ inline v16h load_a_frag(const _Float16* rowbase, int kb) {
    v8h lo = *(const v8h*)(rowbase + kb);
    v8h hi = *(const v8h*)(rowbase + kb + 16);
    v16h a;
#pragma unroll
    for (int i = 0; i < 8; ++i) { a[i] = lo[i]; a[i + 8] = hi[i]; }
    return a;
}
// B stored n-major in LDS (Bt[n][k]): per-lane read = two 8-half runs at kb and kb+16.
__device__ inline v16h load_b_frag(const _Float16* colbase, int kb) {
    v8h lo = *(const v8h*)(colbase + kb);
    v8h hi = *(const v8h*)(colbase + kb + 16);
    v16h b;
#pragma unroll
    for (int i = 0; i < 8; ++i) { b[i] = lo[i]; b[i + 8] = hi[i]; }
    return b;
}
// C/D (16x16 f32): row = (lane>=16 ? 8:0) + r, col = lane&15

// ---------------------------------------------------------------------------
// One K-chunk (32) of the projection GEMM: stage tiles, 4 WMMAs per wave.
// FULL=true: no bounds checks; AF16 path uses async global->LDS b128 copies.
// ---------------------------------------------------------------------------
template <bool AF16, bool FULL>
__device__ __forceinline__ void gemm_chunk(
    const _Float16* __restrict__ A16, const float* __restrict__ A32,
    long rowBase, int lda, int K, int k0,
    const float* __restrict__ W, int colBase,
    _Float16 (*As)[40], _Float16 (*Bt)[40],
    int tid, int lane, int wave, v8f (&acc)[4])
{
    // ---- stage A: 128x32 halves; 2 slots of 8 contiguous K per thread
#pragma unroll
    for (int it = 0; it < 2; ++it) {
        int slot = tid + it * 256;          // 0..511
        int r    = slot >> 2;               // 4 slots per row
        int kq   = (slot & 3) << 3;         // 0,8,16,24
        long gr  = rowBase + r;
        if (FULL && AF16) {
            // 16B/lane global->LDS DMA, tracked by ASYNCcnt (no VGPR round-trip).
            // Generic LDS pointer truncates to the LDS byte offset (ISA 10.2).
            uint32_t lofs = (uint32_t)(uintptr_t)&As[r][kq];
            const _Float16* gsrc = A16 + gr * lda + k0 + kq;
            asm volatile("global_load_async_to_lds_b128 %0, %1, off"
                         :: "v"(lofs), "v"(gsrc) : "memory");
        } else if (FULL) {
            const f32x4* p = (const f32x4*)(A32 + gr * lda + k0 + kq);
            f32x4 f0 = p[0], f1 = p[1];
            v8h hv;
#pragma unroll
            for (int j = 0; j < 4; ++j) {
                hv[j]     = (_Float16)f0[j];
                hv[4 + j] = (_Float16)f1[j];
            }
            *(v8h*)&As[r][kq] = hv;
        } else {
            v8h hv;
#pragma unroll
            for (int j = 0; j < 8; ++j) {
                int kk = k0 + kq + j;
                float v = 0.f;
                if (kk < K)
                    v = AF16 ? (float)A16[gr * lda + kk] : A32[gr * lda + kk];
                hv[j] = (_Float16)v;
            }
            *(v8h*)&As[r][kq] = hv;
        }
    }

    // ---- stage B (n-major): 64x32 halves; 1 slot of 8 per thread (f32->f16)
    {
        int n  = tid >> 2;
        int kq = (tid & 3) << 3;
        const float* wrow = W + (long)(colBase + n) * K + k0 + kq;
        v8h hv;
        if (FULL) {
            f32x4 f0 = ((const f32x4*)wrow)[0];
            f32x4 f1 = ((const f32x4*)wrow)[1];
#pragma unroll
            for (int j = 0; j < 4; ++j) {
                hv[j]     = (_Float16)f0[j];
                hv[4 + j] = (_Float16)f1[j];
            }
        } else {
#pragma unroll
            for (int j = 0; j < 8; ++j) {
                int kk = k0 + kq + j;
                hv[j] = (_Float16)((kk < K) ? wrow[j] : 0.f);
            }
        }
        *(v8h*)&Bt[n][kq] = hv;
    }

    if (FULL && !AF16 && (k0 + 64 <= K)) {
        // prefetch next K-chunk of A while we compute (global_prefetch_b8)
        __builtin_prefetch(A32 + (rowBase + (tid >> 2)) * lda + k0 + 32 + ((tid & 3) << 3), 0, 1);
    }
    if (FULL && AF16)
        asm volatile("s_wait_asynccnt 0x0" ::: "memory");
    __syncthreads();

    v16h a = load_a_frag(&As[wave * 16 + (lane & 15)][0], (lane >> 4) << 3);
#pragma unroll
    for (int nt = 0; nt < 4; ++nt) {
        v16h b = load_b_frag(&Bt[nt * 16 + (lane & 15)][0], (lane >> 4) << 3);
        acc[nt] = __builtin_amdgcn_wmma_f32_16x16x32_f16(
            false, a, false, b, (short)0, acc[nt], false, false);
    }
    __syncthreads();
}

// ---------------------------------------------------------------------------
// out(M x 384, f16) = A(M x K) @ W^T + bias.   W is (384 x K) row-major.
// WG: 256 thr (8 waves), tile 128x64. LDS strides 40 halves -> 16B alignment.
// ---------------------------------------------------------------------------
template <bool AF16>
__global__ __launch_bounds__(256)
void xproj_gemm(const void* __restrict__ Aptr, int lda, int K,
                const float* __restrict__ W, const float* __restrict__ bias,
                _Float16* __restrict__ out)
{
    __shared__ _Float16 As[128][40];    // row-major A tile (128x32 + pad)
    __shared__ _Float16 Bt[64][40];     // n-major B tile  (64x32 + pad)

    const int tid  = threadIdx.x;
    const int lane = tid & 31;
    const int wave = tid >> 5;
    const long rowBase = (long)blockIdx.x * 128;
    const int  colBase = blockIdx.y * 64;

    const _Float16* A16 = (const _Float16*)Aptr;
    const float*    A32 = (const float*)Aptr;

    const v8f vzero = {};
    v8f acc[4];
#pragma unroll
    for (int i = 0; i < 4; ++i) acc[i] = vzero;

    const int kFull = K >> 5;           // full 32-chunks
    for (int kc = 0; kc < kFull; ++kc)
        gemm_chunk<AF16, true>(A16, A32, rowBase, lda, K, kc * 32,
                               W, colBase, As, Bt, tid, lane, wave, acc);
    if (K & 31)
        gemm_chunk<AF16, false>(A16, A32, rowBase, lda, K, kFull * 32,
                                W, colBase, As, Bt, tid, lane, wave, acc);

    const int rsel = (lane >> 4) << 3;
    const int c0   = lane & 15;
#pragma unroll
    for (int nt = 0; nt < 4; ++nt) {
        int col = colBase + nt * 16 + c0;
        float bv = bias[col];
#pragma unroll
        for (int r = 0; r < 8; ++r) {
            long row = rowBase + wave * 16 + rsel + r;
            out[row * G3 + col] = (_Float16)(acc[nt][r] + bv);
        }
    }
}

// ---------------------------------------------------------------------------
// GRU scan: each WG owns 16 batch rows for all 250 steps. 128 thr (4 waves).
// LDS: W_hh^T f16 n-major [384][136] (~102KB resident), h f16 [16][136],
// gh f32 [16][388]. Per step: 96 WMMAs (4 waves x 6 n-tiles x 4 k-chunks).
// ---------------------------------------------------------------------------
#define HSTRIDE 136             // halves; 272B rows, 16B aligned
#define GSTRIDE (G3 + 4)        // 388 floats

template <bool STORE_SEQ, bool STORE_FINAL>
__global__ __launch_bounds__(128)
void gru_scan(const _Float16* __restrict__ xproj,
              const float* __restrict__ W_hh, const float* __restrict__ b_hh,
              _Float16* __restrict__ h_seq_out,   // STORE_SEQ: full h sequence (f16)
              float* __restrict__ h_final_out)    // STORE_FINAL: final h (f32)
{
    extern __shared__ char smem[];
    _Float16* Wt  = (_Float16*)smem;                                 // [384][HSTRIDE]
    _Float16* hsh = (_Float16*)(smem + G3 * HSTRIDE * 2);            // [16][HSTRIDE]
    float*    gsh = (float*)(smem + (G3 + 16) * HSTRIDE * 2);        // [16][GSTRIDE]

    const int tid  = threadIdx.x;
    const int lane = tid & 31;
    const int wave = tid >> 5;
    const int batchBase = blockIdx.x * 16;

    // W_hh is (384 x 128) row-major == n-major B layout; straight f32->f16 copy.
    for (int slot = tid; slot < G3 * 16; slot += 128) {
        int n  = slot >> 4;
        int kq = (slot & 15) << 3;
        const float* wrow = W_hh + (long)n * HIDDEN + kq;
        f32x4 f0 = ((const f32x4*)wrow)[0];
        f32x4 f1 = ((const f32x4*)wrow)[1];
        v8h hv;
#pragma unroll
        for (int j = 0; j < 4; ++j) {
            hv[j]     = (_Float16)f0[j];
            hv[4 + j] = (_Float16)f1[j];
        }
        *(v8h*)(Wt + n * HSTRIDE + kq) = hv;
    }
    {
        const v8h hzero = {};
        for (int slot = tid; slot < 16 * 16; slot += 128) {
            int r = slot >> 4, kq = (slot & 15) << 3;
            *(v8h*)(hsh + r * HSTRIDE + kq) = hzero;
        }
    }
    __syncthreads();

    const v8f vzero = {};
    for (int t = 0; t < SEQ; ++t) {
        // gh = h @ W_hh^T (16 x 384); wave handles cols [wave*96, wave*96+96)
        v8f acc[6];
#pragma unroll
        for (int i = 0; i < 6; ++i) acc[i] = vzero;
#pragma unroll
        for (int kc = 0; kc < 4; ++kc) {
            v16h a = load_a_frag(hsh + (lane & 15) * HSTRIDE, kc * 32 + ((lane >> 4) << 3));
#pragma unroll
            for (int nt = 0; nt < 6; ++nt) {
                const _Float16* colbase =
                    Wt + (wave * 96 + nt * 16 + (lane & 15)) * HSTRIDE;
                v16h b = load_b_frag(colbase, kc * 32 + ((lane >> 4) << 3));
                acc[nt] = __builtin_amdgcn_wmma_f32_16x16x32_f16(
                    false, a, false, b, (short)0, acc[nt], false, false);
            }
        }
        const int rsel = (lane >> 4) << 3;
        const int c0   = lane & 15;
#pragma unroll
        for (int nt = 0; nt < 6; ++nt) {
            int col = wave * 96 + nt * 16 + c0;
            float bv = b_hh[col];
#pragma unroll
            for (int r = 0; r < 8; ++r)
                gsh[(rsel + r) * GSTRIDE + col] = acc[nt][r] + bv;
        }
        __syncthreads();

        // gate math: thread tid = hidden column, loops the 16 batch rows
        for (int row = 0; row < 16; ++row) {
            long grow = (long)(batchBase + row) * SEQ + t;
            const _Float16* xp = xproj + grow * G3;
            float xr = (float)xp[tid];
            float xz = (float)xp[HIDDEN + tid];
            float xn = (float)xp[2 * HIDDEN + tid];
            float hr = gsh[row * GSTRIDE + tid];
            float hz = gsh[row * GSTRIDE + HIDDEN + tid];
            float hn = gsh[row * GSTRIDE + 2 * HIDDEN + tid];
            float hp = (float)hsh[row * HSTRIDE + tid];
            float rg = 1.f / (1.f + __expf(-(xr + hr)));
            float zg = 1.f / (1.f + __expf(-(xz + hz)));
            float ng = tanhf(xn + rg * hn);
            float hnew = (1.f - zg) * ng + zg * hp;
            hsh[row * HSTRIDE + tid] = (_Float16)hnew;
            if (STORE_SEQ)
                h_seq_out[grow * HIDDEN + tid] = (_Float16)hnew;
            if (STORE_FINAL && t == SEQ - 1)
                h_final_out[(long)(batchBase + row) * HIDDEN + tid] = hnew;
        }
        __syncthreads();
    }
}

// ---------------------------------------------------------------------------
// MLP head: y = silu(LN(h@w1^T+b1)) @ w2^T + b2.  One WG (128 thr) per row.
// ---------------------------------------------------------------------------
__global__ __launch_bounds__(128)
void mlp_head_kernel(const float* __restrict__ h_t,
                     const float* __restrict__ w1, const float* __restrict__ b1,
                     const float* __restrict__ g,  const float* __restrict__ beta,
                     const float* __restrict__ w2, const float* __restrict__ b2,
                     float* __restrict__ out)
{
    __shared__ float hsh[HIDDEN];
    __shared__ float ysh[HIDDEN];
    __shared__ float red[128];
    int tid = threadIdx.x;
    int row = blockIdx.x;

    hsh[tid] = h_t[(long)row * HIDDEN + tid];
    __syncthreads();

    float acc = b1[tid];
    const float* wr = w1 + (long)tid * HIDDEN;
#pragma unroll 8
    for (int k = 0; k < HIDDEN; ++k) acc += hsh[k] * wr[k];

    red[tid] = acc; __syncthreads();
    for (int s = 64; s > 0; s >>= 1) { if (tid < s) red[tid] += red[tid + s]; __syncthreads(); }
    float mu = red[0] / HIDDEN;
    __syncthreads();
    float d = acc - mu;
    red[tid] = d * d; __syncthreads();
    for (int s = 64; s > 0; s >>= 1) { if (tid < s) red[tid] += red[tid + s]; __syncthreads(); }
    float var = red[0] / HIDDEN;

    float y = d * rsqrtf(var + LN_EPS_c) * g[tid] + beta[tid];
    y = y / (1.f + __expf(-y));         // y * sigmoid(y)
    ysh[tid] = y;
    __syncthreads();

    for (int s2 = tid; s2 < N_STOCKS; s2 += 128) {
        float a2 = b2[s2];
        const float* w2r = w2 + (long)s2 * HIDDEN;
#pragma unroll 8
        for (int k = 0; k < HIDDEN; ++k) a2 += ysh[k] * w2r[k];
        out[(long)row * N_STOCKS + s2] = a2;
    }
}

// ---------------------------------------------------------------------------
// top-50 (sigmoid monotone -> select on raw att) + masked softmax +
// 32-iter clip rebalance. One WG (256 thr) per batch row, LDS reductions.
// ---------------------------------------------------------------------------
__global__ __launch_bounds__(256)
void select_rebalance(const float* __restrict__ att, const float* __restrict__ scores,
                      float* __restrict__ out)
{
    __shared__ float a[N_STOCKS];
    __shared__ float w[N_STOCKS];
    __shared__ float m[N_STOCKS];
    __shared__ float redv[256];
    __shared__ int   redi[256];
    __shared__ int   doneFlag;
    int tid = threadIdx.x;
    int row = blockIdx.x;

    for (int i = tid; i < N_STOCKS; i += 256) {
        a[i] = att[(long)row * N_STOCKS + i];
        m[i] = 0.f;
    }
    __syncthreads();

    // top-50 via iterative argmax
    for (int it = 0; it < NSEL; ++it) {
        float bv = -__builtin_inff(); int bi = 0;
        for (int i = tid; i < N_STOCKS; i += 256)
            if (a[i] > bv) { bv = a[i]; bi = i; }
        redv[tid] = bv; redi[tid] = bi;
        __syncthreads();
        for (int s = 128; s > 0; s >>= 1) {
            if (tid < s && redv[tid + s] > redv[tid]) { redv[tid] = redv[tid + s]; redi[tid] = redi[tid + s]; }
            __syncthreads();
        }
        if (tid == 0) { int j = redi[0]; m[j] = 1.f; a[j] = -__builtin_inff(); }
        __syncthreads();
    }

    // masked softmax + normalize
    float lm = -__builtin_inff();
    for (int i = tid; i < N_STOCKS; i += 256) { float v = scores[(long)row * N_STOCKS + i]; w[i] = v; lm = fmaxf(lm, v); }
    redv[tid] = lm; __syncthreads();
    for (int s = 128; s > 0; s >>= 1) { if (tid < s) redv[tid] = fmaxf(redv[tid], redv[tid + s]); __syncthreads(); }
    float mx = redv[0]; __syncthreads();

    float ls = 0.f;
    for (int i = tid; i < N_STOCKS; i += 256) { float e = __expf(w[i] - mx); w[i] = e; ls += e; }
    redv[tid] = ls; __syncthreads();
    for (int s = 128; s > 0; s >>= 1) { if (tid < s) redv[tid] += redv[tid + s]; __syncthreads(); }
    float denom = redv[0]; __syncthreads();

    ls = 0.f;
    for (int i = tid; i < N_STOCKS; i += 256) { float v = (w[i] / denom) * m[i]; w[i] = v; ls += v; }
    redv[tid] = ls; __syncthreads();
    for (int s = 128; s > 0; s >>= 1) { if (tid < s) redv[tid] += redv[tid + s]; __syncthreads(); }
    float wsum = redv[0] + 1e-8f; __syncthreads();

    for (int i = tid; i < N_STOCKS; i += 256) {
        float v = w[i] / wsum;
        float sel = (v > 0.f) ? 1.f : 0.f;
        m[i] = sel;
        w[i] = fminf(fmaxf(sel > 0.f ? v : 0.f, LBc), UBc);
    }
    if (tid == 0) doneFlag = 0;
    __syncthreads();

    for (int it = 0; it < REBAL_IT; ++it) {
        float s1 = 0.f;
        for (int i = tid; i < N_STOCKS; i += 256) s1 += w[i];
        redv[tid] = s1; __syncthreads();
        for (int s = 128; s > 0; s >>= 1) { if (tid < s) redv[tid] += redv[tid + s]; __syncthreads(); }
        float excess = redv[0] - 1.f;
        __syncthreads();
        int pos = excess > 0.f;

        float cnt = 0.f;
        for (int i = tid; i < N_STOCKS; i += 256) {
            float wi = w[i];
            if (((pos ? (wi > LBc) : (wi < UBc))) && m[i] > 0.f) cnt += 1.f;
        }
        redv[tid] = cnt; __syncthreads();
        for (int s = 128; s > 0; s >>= 1) { if (tid < s) redv[tid] += redv[tid + s]; __syncthreads(); }
        float n_adj = redv[0];
        __syncthreads();

        int done   = doneFlag;
        int active = (fabsf(excess) > 1e-6f) && !done;
        int stuck  = (n_adj == 0.f);
        float step = excess / fmaxf(n_adj, 1.f);
        if (active && !stuck) {
            for (int i = tid; i < N_STOCKS; i += 256) {
                float wi = w[i];
                if (((pos ? (wi > LBc) : (wi < UBc))) && m[i] > 0.f)
                    w[i] = fminf(fmaxf(wi - step, LBc), UBc);
            }
        }
        __syncthreads();
        if (tid == 0 && active && stuck) doneFlag = 1;
        __syncthreads();
    }

    for (int i = tid; i < N_STOCKS; i += 256)
        out[(long)row * N_STOCKS + i] = w[i];
}

// ---------------------------------------------------------------------------
extern "C" void kernel_launch(void* const* d_in, const int* in_sizes, int n_in,
                              void* d_out, int out_size, void* d_ws, size_t ws_size,
                              hipStream_t stream)
{
    const float* x      = (const float*)d_in[0];
    const float* W_ih0  = (const float*)d_in[1];
    const float* W_hh0  = (const float*)d_in[2];
    const float* b_ih0  = (const float*)d_in[3];
    const float* b_hh0  = (const float*)d_in[4];
    const float* W_ih1  = (const float*)d_in[5];
    const float* W_hh1  = (const float*)d_in[6];
    const float* b_ih1  = (const float*)d_in[7];
    const float* b_hh1  = (const float*)d_in[8];
    const float* att_w1 = (const float*)d_in[9];
    const float* att_b1 = (const float*)d_in[10];
    const float* att_g  = (const float*)d_in[11];
    const float* att_be = (const float*)d_in[12];
    const float* att_w2 = (const float*)d_in[13];
    const float* att_b2 = (const float*)d_in[14];
    const float* sc_w1  = (const float*)d_in[15];
    const float* sc_b1  = (const float*)d_in[16];
    const float* sc_g   = (const float*)d_in[17];
    const float* sc_be  = (const float*)d_in[18];
    const float* sc_w2  = (const float*)d_in[19];
    const float* sc_b2  = (const float*)d_in[20];

    // workspace carving (~133 MB): xproj f16 buffer reused by both layers
    char* ws = (char*)d_ws;
    size_t off = 0;
    auto carve = [&](size_t bytes) {
        void* p = ws + off;
        off = (off + bytes + 255) & ~(size_t)255;
        return p;
    };
    _Float16* xproj = (_Float16*)carve((size_t)M_ROWS * G3 * sizeof(_Float16));     // 98.3 MB
    _Float16* hseq  = (_Float16*)carve((size_t)M_ROWS * HIDDEN * sizeof(_Float16)); // 32.8 MB
    float*    h_t   = (float*)carve((size_t)BATCH * HIDDEN * sizeof(float));
    float*    attv  = (float*)carve((size_t)BATCH * N_STOCKS * sizeof(float));
    float*    scv   = (float*)carve((size_t)BATCH * N_STOCKS * sizeof(float));

    dim3 gGemm(M_ROWS / 128, G3 / 64);   // 1000 x 6 WGs
    size_t smemScan = (size_t)(G3 + 16) * HSTRIDE * sizeof(_Float16)
                    + (size_t)16 * GSTRIDE * sizeof(float); // ~133.6 KB dynamic LDS

    // layer 0: x-projection (fp32 A, K=500: 15 full chunks + tail), then scan
    xproj_gemm<false><<<gGemm, 256, 0, stream>>>(x, N_STOCKS, N_STOCKS,
                                                 W_ih0, b_ih0, xproj);
    gru_scan<true, false><<<BATCH / 16, 128, smemScan, stream>>>(
        xproj, W_hh0, b_hh0, hseq, nullptr);

    // layer 1: projection of layer-0 hidden sequence (f16 A, K=128, async LDS)
    xproj_gemm<true><<<gGemm, 256, 0, stream>>>(hseq, HIDDEN, HIDDEN,
                                                W_ih1, b_ih1, xproj);
    gru_scan<false, true><<<BATCH / 16, 128, smemScan, stream>>>(
        xproj, W_hh1, b_hh1, nullptr, h_t);

    // heads + selection + rebalance
    mlp_head_kernel<<<BATCH, 128, 0, stream>>>(h_t, att_w1, att_b1, att_g, att_be,
                                               att_w2, att_b2, attv);
    mlp_head_kernel<<<BATCH, 128, 0, stream>>>(h_t, sc_w1, sc_b1, sc_g, sc_be,
                                               sc_w2, sc_b2, scv);
    select_rebalance<<<BATCH, 256, 0, stream>>>(attv, scv, (float*)d_out);
}
